// Attention_83348135346609
// MI455X (gfx1250) — matrix operand order repaired and checked by reading, not compiled, not run on hardware
//
#include <hip/hip_runtime.h>
#include <hip/hip_bf16.h>

// ---------------------------------------------------------------------------
// Attention block for MI455X (gfx1250, wave32, WMMA f32_16x16x32_f16)
//   x[8,512,32,32] -> qkv (1x1 conv) -> per-head attention (softmax over keys)
//   -> 1x1 conv proj + bias -> out[8,512,32,32] (f32)
// ---------------------------------------------------------------------------

typedef __attribute__((ext_vector_type(16))) _Float16 v16h;
typedef __attribute__((ext_vector_type(8)))  _Float16 v8h;
typedef __attribute__((ext_vector_type(8)))  float    v8f;

union Frag16 { v16h v; v8h h[2]; };

static constexpr int B_  = 8;
static constexpr int C_  = 512;
static constexpr int N_  = 1024;   // H*W
static constexpr int NH_ = 8;
static constexpr int HD_ = 64;
static constexpr int O3_ = 1536;   // 3*C

static __device__ __forceinline__ v8h load8(const _Float16* p) {
  return *(const v8h*)p;
}
static __device__ __forceinline__ v8f vzero8() {
  v8f z = {0.f,0.f,0.f,0.f,0.f,0.f,0.f,0.f};
  return z;
}
// exchange with lane^16 (wave32): ds_swizzle SWAPX16 (xor=0x10,and=0x1f)
static __device__ __forceinline__ float swapx16(float x) {
  int i = __float_as_int(x);
  i = __builtin_amdgcn_ds_swizzle(i, 0x401F);
  return __int_as_float(i);
}
static __device__ __forceinline__ v8f wmma_f16(const Frag16& a, const Frag16& b, v8f c) {
  return __builtin_amdgcn_wmma_f32_16x16x32_f16(false, a.v, false, b.v,
                                                (short)0, c, false, false);
}

// ------------------------------ converts -----------------------------------

__global__ void cvt_f32_f16_kernel(const float* __restrict__ src,
                                   _Float16* __restrict__ dst, int n) {
  int i = blockIdx.x * blockDim.x + threadIdx.x;
  if (i < n) dst[i] = (_Float16)src[i];
}

// x: [B][C][N] f32  ->  xT: [B][N][C] f16   (coalesced read over n)
__global__ void transpose_cvt_kernel(const float* __restrict__ x,
                                     _Float16* __restrict__ xT) {
  int i = blockIdx.x * blockDim.x + threadIdx.x;
  if (i >= B_ * C_ * N_) return;
  int n = i % N_;
  int t = i / N_;
  int c = t % C_;
  int b = t / C_;
  xT[(size_t)b * N_ * C_ + (size_t)n * C_ + c] = (_Float16)x[i];
}

// ------------------------------ QKV GEMM -----------------------------------
// qkvT[b][n][o] = sum_c W[o][c] * xT[b][n][c]     (M=1536, N=1024, K=512)
// Block 256 thr (8 waves, 2x4), block tile 128x128, wave tile 64x32.
__global__ __launch_bounds__(256) void qkv_gemm_kernel(
    const _Float16* __restrict__ W,    // [1536][512]
    const _Float16* __restrict__ xT,   // [B][1024][512]
    _Float16* __restrict__ qkvT)       // [B][1024][1536]
{
  const int lane = threadIdx.x & 31;
  const int wave = threadIdx.x >> 5;
  const int lcol = lane & 15;
  const int lhi  = lane >> 4;
  const int o0 = blockIdx.x * 128 + (wave >> 2) * 64;
  const int n0 = blockIdx.y * 128 + (wave & 3) * 32;
  const int b  = blockIdx.z;

  const _Float16* xb = xT + (size_t)b * N_ * C_;
  _Float16* qb = qkvT + (size_t)b * N_ * O3_;

  v8f acc[4][2];
#pragma unroll
  for (int i = 0; i < 4; ++i)
#pragma unroll
    for (int j = 0; j < 2; ++j) acc[i][j] = vzero8();

  for (int cc = 0; cc < C_; cc += 32) {
    Frag16 a[4], bf[2];
#pragma unroll
    for (int mt = 0; mt < 4; ++mt) {
      const _Float16* ap = W + (size_t)(o0 + mt * 16 + lcol) * C_ + cc + 8 * lhi;
      a[mt].h[0] = load8(ap);
      a[mt].h[1] = load8(ap + 16);
    }
#pragma unroll
    for (int nt = 0; nt < 2; ++nt) {
      const _Float16* bp = xb + (size_t)(n0 + nt * 16 + lcol) * C_ + cc + 8 * lhi;
      bf[nt].h[0] = load8(bp);
      bf[nt].h[1] = load8(bp + 16);
    }
#pragma unroll
    for (int mt = 0; mt < 4; ++mt)
#pragma unroll
      for (int nt = 0; nt < 2; ++nt)
        acc[mt][nt] = wmma_f16(a[mt], bf[nt], acc[mt][nt]);
  }

#pragma unroll
  for (int mt = 0; mt < 4; ++mt)
#pragma unroll
    for (int nt = 0; nt < 2; ++nt) {
      v8h o;
#pragma unroll
      for (int j = 0; j < 8; ++j) o[j] = (_Float16)acc[mt][nt][j];
      const int n = n0 + nt * 16 + lcol;
      const int orow = o0 + mt * 16 + lhi * 8;   // 8 contiguous o channels
      *(v8h*)(qb + (size_t)n * O3_ + orow) = o;
    }
}

// --------------------------- flash attention -------------------------------
// per (b,h): S = K^T Q * scale, softmax over keys, O = V * P
// qkvT layout: q at o=h*192+d, k at +64, v at +128, channel-contiguous per n.
// Grid (64 bh, 16 qblocks), 128 threads = 4 waves; each wave owns 16 queries.
__global__ __launch_bounds__(128) void attn_kernel(
    const _Float16* __restrict__ qkvT,   // [B][1024][1536]
    _Float16* __restrict__ outT)         // [B][1024][512]
{
  __shared__ _Float16 Vlds[HD_ * 64];       // [d][nk]  64x64  (8 KB)
  __shared__ _Float16 Plds[4 * 16 * 64];    // per-wave [nq16][nk64] (8 KB)

  const int lane = threadIdx.x & 31;
  const int wave = threadIdx.x >> 5;
  const int lcol = lane & 15;
  const int lhi  = lane >> 4;
  const int b = blockIdx.x >> 3;
  const int h = blockIdx.x & 7;
  const int nq = blockIdx.y * 64 + wave * 16 + lcol;   // this lane's query col

  const _Float16* base = qkvT + (size_t)b * N_ * O3_;
  const int qoff = h * 192;
  const int koff = h * 192 + 64;
  const int voff = h * 192 + 128;

  // Q B-frags (K=d, 2 chunks of 32), pre-scaled by hd^-0.5 = 0.125 (exact)
  Frag16 qf[2];
#pragma unroll
  for (int dc = 0; dc < 2; ++dc) {
    const _Float16* qp = base + (size_t)nq * O3_ + qoff + dc * 32 + 8 * lhi;
    qf[dc].h[0] = load8(qp);
    qf[dc].h[1] = load8(qp + 16);
#pragma unroll
    for (int i = 0; i < 16; ++i) qf[dc].v[i] = qf[dc].v[i] * (_Float16)0.125f;
  }

  v8f oacc[4];
#pragma unroll
  for (int dt = 0; dt < 4; ++dt) oacc[dt] = vzero8();
  float m_run = -__builtin_inff();
  float l_run = 0.f;

  _Float16* PldsW = Plds + wave * 16 * 64;

  for (int kb = 0; kb < N_; kb += 64) {
    // ---- cooperative stage of V block transposed: Vlds[d][nk] ----
    {
      const int nk = threadIdx.x >> 1;
      const int dh = (threadIdx.x & 1) * 32;
      const _Float16* vp = base + (size_t)(kb + nk) * O3_ + voff + dh;
#pragma unroll
      for (int j = 0; j < 4; ++j) {
        v8h v = load8(vp + 8 * j);
#pragma unroll
        for (int e = 0; e < 8; ++e) Vlds[(dh + 8 * j + e) * 64 + nk] = v[e];
      }
    }
    __syncthreads();

    // ---- S tiles: 4 key tiles x 16 queries, K-frags direct from global ----
    v8f s[4];
#pragma unroll
    for (int t = 0; t < 4; ++t) {
      const _Float16* kp = base + (size_t)(kb + t * 16 + lcol) * O3_ + koff;
      Frag16 ka0, ka1;
      ka0.h[0] = load8(kp + 8 * lhi);
      ka0.h[1] = load8(kp + 8 * lhi + 16);
      ka1.h[0] = load8(kp + 32 + 8 * lhi);
      ka1.h[1] = load8(kp + 32 + 8 * lhi + 16);
      v8f z = wmma_f16(ka0, qf[0], vzero8());
      s[t] = wmma_f16(ka1, qf[1], z);
    }

    // ---- online softmax over keys (rows of D-tiles) ----
    float mb = -__builtin_inff();
#pragma unroll
    for (int t = 0; t < 4; ++t)
#pragma unroll
      for (int j = 0; j < 8; ++j) mb = fmaxf(mb, s[t][j]);
    mb = fmaxf(mb, swapx16(mb));                  // combine lane n <-> n+16
    const float m_new = fmaxf(m_run, mb);
    const float alpha = __expf(m_run - m_new);
    float lb = 0.f;
#pragma unroll
    for (int t = 0; t < 4; ++t) {
      v8h p;
#pragma unroll
      for (int j = 0; j < 8; ++j) {
        float e = __expf(s[t][j] - m_new);
        lb += e;
        p[j] = (_Float16)e;
      }
      // D-layout row-run is contiguous in Plds[nq][nk] -> single b128 store
      *(v8h*)(PldsW + lcol * 64 + t * 16 + lhi * 8) = p;
    }
    lb += swapx16(lb);
    l_run = l_run * alpha + lb;
    m_run = m_new;
#pragma unroll
    for (int dt = 0; dt < 4; ++dt)
#pragma unroll
      for (int j = 0; j < 8; ++j) oacc[dt][j] *= alpha;

    // ---- O += V * P ----
    Frag16 pb[2];
#pragma unroll
    for (int kc = 0; kc < 2; ++kc) {
      const _Float16* pp = PldsW + lcol * 64 + kc * 32 + 8 * lhi;
      pb[kc].h[0] = *(const v8h*)pp;
      pb[kc].h[1] = *(const v8h*)(pp + 16);
    }
#pragma unroll
    for (int dt = 0; dt < 4; ++dt)
#pragma unroll
      for (int kc = 0; kc < 2; ++kc) {
        const _Float16* vp2 = &Vlds[(dt * 16 + lcol) * 64 + kc * 32 + 8 * lhi];
        Frag16 va;
        va.h[0] = *(const v8h*)vp2;
        va.h[1] = *(const v8h*)(vp2 + 16);
        oacc[dt] = wmma_f16(va, pb[kc], oacc[dt]);
      }
    __syncthreads();
  }

  // ---- normalize and store: outT[b][n][c], c contiguous per lane ----
  const float inv = 1.0f / l_run;
  _Float16* ob = outT + (size_t)b * N_ * C_ + (size_t)nq * C_;
#pragma unroll
  for (int dt = 0; dt < 4; ++dt) {
    v8h o;
#pragma unroll
    for (int j = 0; j < 8; ++j) o[j] = (_Float16)(oacc[dt][j] * inv);
    const int c0 = h * 64 + dt * 16 + lhi * 8;
    *(v8h*)(ob + c0) = o;
  }
}

// ----------------------------- proj GEMM -----------------------------------
// out[b][o][n] = sum_c Wp[o][c] * attT[b][n][c] + bias[o]   (f32 output)
__global__ __launch_bounds__(256) void proj_gemm_kernel(
    const _Float16* __restrict__ W,    // [512][512]
    const _Float16* __restrict__ aT,   // [B][1024][512]
    const float* __restrict__ bias,    // [512]
    float* __restrict__ out)           // [B][512][1024]
{
  const int lane = threadIdx.x & 31;
  const int wave = threadIdx.x >> 5;
  const int lcol = lane & 15;
  const int lhi  = lane >> 4;
  const int o0 = blockIdx.x * 128 + (wave >> 2) * 64;
  const int n0 = blockIdx.y * 128 + (wave & 3) * 32;
  const int b  = blockIdx.z;

  const _Float16* ab = aT + (size_t)b * N_ * C_;
  float* ob = out + (size_t)b * C_ * N_;

  v8f acc[4][2];
#pragma unroll
  for (int i = 0; i < 4; ++i)
#pragma unroll
    for (int j = 0; j < 2; ++j) acc[i][j] = vzero8();

  for (int cc = 0; cc < C_; cc += 32) {
    Frag16 a[4], bf[2];
#pragma unroll
    for (int mt = 0; mt < 4; ++mt) {
      const _Float16* ap = W + (size_t)(o0 + mt * 16 + lcol) * C_ + cc + 8 * lhi;
      a[mt].h[0] = load8(ap);
      a[mt].h[1] = load8(ap + 16);
    }
#pragma unroll
    for (int nt = 0; nt < 2; ++nt) {
      const _Float16* bp = ab + (size_t)(n0 + nt * 16 + lcol) * C_ + cc + 8 * lhi;
      bf[nt].h[0] = load8(bp);
      bf[nt].h[1] = load8(bp + 16);
    }
#pragma unroll
    for (int mt = 0; mt < 4; ++mt)
#pragma unroll
      for (int nt = 0; nt < 2; ++nt)
        acc[mt][nt] = wmma_f16(a[mt], bf[nt], acc[mt][nt]);
  }

#pragma unroll
  for (int mt = 0; mt < 4; ++mt)
#pragma unroll
    for (int nt = 0; nt < 2; ++nt) {
      const int n = n0 + nt * 16 + lcol;
#pragma unroll
      for (int j = 0; j < 8; ++j) {
        const int o = o0 + mt * 16 + lhi * 8 + j;   // lanes coalesce over n
        ob[(size_t)o * N_ + n] = acc[mt][nt][j] + bias[o];
      }
    }
}

// ------------------------------ launcher -----------------------------------

extern "C" void kernel_launch(void* const* d_in, const int* in_sizes, int n_in,
                              void* d_out, int out_size, void* d_ws, size_t ws_size,
                              hipStream_t stream) {
  const float* x      = (const float*)d_in[0];   // [8,512,32,32]
  const float* w_qkv  = (const float*)d_in[1];   // [1536,512]
  const float* w_proj = (const float*)d_in[2];   // [512,512]
  const float* b_proj = (const float*)d_in[3];   // [512]
  float* out = (float*)d_out;                    // [8,512,32,32]

  char* ws = (char*)d_ws;
  size_t off = 0;
  auto carve = [&](size_t bytes) -> void* {
    off = (off + 255) & ~(size_t)255;
    void* p = ws + off;
    off += bytes;
    return p;
  };
  _Float16* wqkv_h = (_Float16*)carve((size_t)O3_ * C_ * 2);       // 1.5 MB
  _Float16* wproj_h = (_Float16*)carve((size_t)C_ * C_ * 2);       // 0.5 MB
  _Float16* xT      = (_Float16*)carve((size_t)B_ * N_ * C_ * 2);  // 8 MB
  _Float16* qkvT    = (_Float16*)carve((size_t)B_ * N_ * O3_ * 2); // 24 MB
  _Float16* attT    = (_Float16*)carve((size_t)B_ * N_ * C_ * 2);  // 8 MB
  (void)ws_size; (void)in_sizes; (void)n_in; (void)out_size;

  {
    int n = O3_ * C_;
    cvt_f32_f16_kernel<<<(n + 255) / 256, 256, 0, stream>>>(w_qkv, wqkv_h, n);
  }
  {
    int n = C_ * C_;
    cvt_f32_f16_kernel<<<(n + 255) / 256, 256, 0, stream>>>(w_proj, wproj_h, n);
  }
  {
    int n = B_ * C_ * N_;
    transpose_cvt_kernel<<<(n + 255) / 256, 256, 0, stream>>>(x, xT);
  }
  qkv_gemm_kernel<<<dim3(O3_ / 128, N_ / 128, B_), 256, 0, stream>>>(wqkv_h, xT, qkvT);
  attn_kernel<<<dim3(B_ * NH_, N_ / 64), 128, 0, stream>>>(qkvT, attT);
  proj_gemm_kernel<<<dim3(C_ / 128, N_ / 128, B_), 256, 0, stream>>>(wproj_h, attT, b_proj, out);
}
